// DepthModuleDotMulti_88880053223989
// MI455X (gfx1250) — matrix operand order repaired and checked by reading, not compiled, not run on hardware
//
#include <hip/hip_runtime.h>
#include <hip/hip_bf16.h>

typedef __attribute__((ext_vector_type(4)))  __bf16 v4bf;   // 8B
typedef __attribute__((ext_vector_type(8)))  __bf16 v8bf;   // 16B
typedef __attribute__((ext_vector_type(16))) __bf16 v16bf;  // 32B
typedef __attribute__((ext_vector_type(8)))  float  v8f;

union Frag16 { v8bf h[2]; v16bf v; };

#define B_      2
#define M_      2
#define F_      6
#define C_      32
#define H_      64
#define W_      64
#define D_      32
#define GROUPS_ 8
#define BASE_   8
#define K_SAMP  (B_*F_)          // 12
#define PLANE   (H_*W_)          // 4096
#define VOL     (D_*PLANE)       // 131072
#define KSTEPS  7                // 224 padded K = 28 taps * 8 cin
#define LDS_X   68               // x slots: 1 + 64 + 3 pad
#define ROWE    (LDS_X*8)        // elements per (plane,kh) row = 544
#define DCHUNK  16
#define NPLANE  (DCHUNK+2)       // 18 input planes per chunk
#define DLBYTES (3*LDS_X*8*2)    // byte stride per +1 output d  = 3264
#define ZOFFB   (66*8*2)         // byte offset of an always-zero 16B LDS slot

// ---------------------------------------------------------------------------
// Stage 1: per-view scalars -> cos(phase)[b][d]
// ---------------------------------------------------------------------------
__global__ void phase_kernel(const float* __restrict__ cam,
                             const float* __restrict__ tw,
                             const float* __restrict__ hypo,
                             float* __restrict__ cosphi) {
  int t = threadIdx.x;
  if (t >= B_ * D_) return;
  int b = t / D_, d = t % D_;
  float sc = 0.f;
  for (int i = 0; i < 9; ++i)  sc += cam[(b*M_ + 1)*9 + i];
  sc = tanhf(sc / 9.f);
  float st = 0.f;
  for (int i = 0; i < 16; ++i) st += tw[(b*M_ + 1)*16 + i];
  st = tanhf(st / 16.f);
  cosphi[t] = cosf(sc * hypo[b*D_ + d] + st);
}

// ---------------------------------------------------------------------------
// Stage 2: combined[k][g][d][y][x] = cosphi[b][d] * (1/4) sum_cc ref*src (bf16)
// ---------------------------------------------------------------------------
__global__ __launch_bounds__(256)
void combine_kernel(const float* __restrict__ feat,
                    const float* __restrict__ cosphi,
                    __bf16* __restrict__ comb) {
  int idx = blockIdx.x * blockDim.x + threadIdx.x;     // (k, g, y, x)
  if (idx >= K_SAMP * GROUPS_ * PLANE) return;
  int xy = idx & (PLANE - 1);
  int g  = (idx >> 12) & (GROUPS_ - 1);
  int k  = idx >> 15;
  int b = k / F_, f = k % F_;
  size_t base0 = (((size_t)(b*M_ + 0)*F_ + f)*C_ + g*4) * PLANE + xy;
  size_t base1 = (((size_t)(b*M_ + 1)*F_ + f)*C_ + g*4) * PLANE + xy;
  float S = 0.f;
  #pragma unroll
  for (int cc = 0; cc < 4; ++cc)
    S += feat[base0 + (size_t)cc*PLANE] * feat[base1 + (size_t)cc*PLANE];
  S *= 0.25f;
  size_t ob = (size_t)(k*GROUPS_ + g) * VOL + xy;
  #pragma unroll
  for (int d = 0; d < D_; ++d)
    comb[ob + (size_t)d*PLANE] = (__bf16)(S * cosphi[b*D_ + d]);
}

// ---------------------------------------------------------------------------
// Stage 3: 8->COUT 3x3x3 conv3d (SAME) + bias (+ReLU) via bf16 WMMA.
//   A = patches (M = 16 x-positions), B = weights (N = cout), K = 224.
//   Halo LDS layout: [plane 18][kh 3][xloc 68][cin 8] (cin innermost ->
//   every fragment 8-run is one aligned ds_load_b128; invalid taps read the
//   always-zero pad slot so no exec masking is needed).
// ---------------------------------------------------------------------------
template<int COUT, bool RELU, typename OUT_T>
__global__ __launch_bounds__(128)
void conv3d_wmma_kernel(const __bf16* __restrict__ X,
                        const float*  __restrict__ Wt,
                        const float*  __restrict__ Bs,
                        OUT_T* __restrict__ Y) {
  __shared__ __bf16 halo[NPLANE * 3 * LDS_X * 8];      // 29376 el = 57.4 KB
  const int tid  = threadIdx.x;
  const int lane = tid & 31;
  const int wave = tid >> 5;                           // 4 waves: x0 = wave*16
  const int x0   = wave * 16;

  // ---- Weight B-fragments: lane = cout column, K 0-15 / 16-31 per half ----
  v16bf bw[KSTEPS];
  const int co16 = lane & 15;
  {
    const int bk0 = (lane < 16) ? 0 : 16;
    #pragma unroll
    for (int s = 0; s < KSTEPS; ++s) {
      v16bf wv;
      #pragma unroll
      for (int j = 0; j < 16; ++j) {
        int kg  = s*32 + bk0 + j;
        int tap = kg >> 3;
        int cin = kg & 7;
        float w = 0.f;
        if (tap < 27 && co16 < COUT) {
          int kd = tap/9, kh = (tap%9)/3, kw = tap%3;
          w = Wt[(((co16*GROUPS_ + cin)*3 + kd)*3 + kh)*3 + kw];
        }
        wv[j] = (__bf16)w;
      }
      bw[s] = wv;
    }
  }
  const float bias = Bs[co16 < COUT ? co16 : 0];

  // ---- Per-lane A-fragment byte offsets into halo (d-part folds into the
  //      ds-instruction immediate: +dl*DLBYTES).
  const int m    = lane & 15;
  const int xme  = x0 + m;
  const int aSel = (lane < 16) ? 0 : 1;
  const int xoff = (lane < 16) ? 0 : 8;                // D-matrix x offset
  int adL[KSTEPS], adH[KSTEPS];
  #pragma unroll
  for (int s = 0; s < KSTEPS; ++s) {
    int tl = s*4 + 2*aSel;                             // lo tap (< 27 always)
    int th = tl + 1;                                   // hi tap
    int kd = tl/9, kh = (tl%9)/3, kw = tl%3;
    adL[s] = (((kd*3 + kh)*LDS_X + xme + kw) * 8) * 2;
    if (th < 27) {
      kd = th/9; kh = (th%9)/3; kw = th%3;
      adH[s] = (((kd*3 + kh)*LDS_X + xme + kw) * 8) * 2;
    } else {
      adH[s] = ZOFFB;                                  // always-zero slot
    }
  }

  const int row = blockIdx.x;                          // 0 .. K_SAMP*H_-1
  if (row >= K_SAMP * H_) return;
  const int y = row % H_;
  const int k = row / H_;
  const char* hb = (const char*)halo;

  for (int dc = 0; dc < 2; ++dc) {
    __syncthreads();
    // ---- edge zeros: xloc 0 and 65..67, all cin, every (plane,kh) row ----
    for (int e = tid; e < NPLANE*3; e += 128) {
      __bf16* rp = halo + e*ROWE;
      v8bf z = {};
      *(v8bf*)(rp)          = z;
      *(v8bf*)(rp + 65*8)   = z;
      *(v8bf*)(rp + 66*8)   = z;
      *(v8bf*)(rp + 67*8)   = z;
    }
    // ---- main fill: group = (plane, kh, xg); 4 x * 8 cin per group ----
    for (int g = tid; g < NPLANE*3*16; g += 128) {
      int xg = g & 15;
      int r  = g >> 4;
      int kh = r % 3;
      int p  = r / 3;                                  // 0..17
      int dd = dc*DCHUNK - 1 + p;
      int yy = y + kh - 1;
      bool inR = ((unsigned)dd < D_) & ((unsigned)yy < H_);
      v4bf q0={},q1={},q2={},q3={},q4={},q5={},q6={},q7={};
      if (inR) {
        const __bf16* s0 = X + ((size_t)k*GROUPS_)*VOL
                             + (size_t)dd*PLANE + yy*W_ + xg*4;
        q0 = *(const v4bf*)(s0);
        q1 = *(const v4bf*)(s0 + VOL);
        q2 = *(const v4bf*)(s0 + 2*(size_t)VOL);
        q3 = *(const v4bf*)(s0 + 3*(size_t)VOL);
        q4 = *(const v4bf*)(s0 + 4*(size_t)VOL);
        q5 = *(const v4bf*)(s0 + 5*(size_t)VOL);
        q6 = *(const v4bf*)(s0 + 6*(size_t)VOL);
        q7 = *(const v4bf*)(s0 + 7*(size_t)VOL);
      }
      int db = ((p*3 + kh)*LDS_X + xg*4 + 1) * 8;      // xloc = 4*xg+1 .. +4
      #pragma unroll
      for (int q = 0; q < 4; ++q) {
        v8bf t = { q0[q], q1[q], q2[q], q3[q], q4[q], q5[q], q6[q], q7[q] };
        *(v8bf*)(halo + db + q*8) = t;                 // ds_store_b128
      }
    }
    __syncthreads();

    // ---- compute 16 d-rows; dl fully unrolled so +dl*DLBYTES becomes the
    //      ds_load immediate offset (no per-iteration address VALU) ----
    #pragma unroll
    for (int dl = 0; dl < DCHUNK; ++dl) {
      const int d = dc*DCHUNK + dl;
      v8f c = {};
      #pragma unroll
      for (int s = 0; s < KSTEPS; ++s) {
        Frag16 a;
        a.h[0] = *(const v8bf*)(hb + adL[s] + dl*DLBYTES);
        a.h[1] = *(const v8bf*)(hb + adH[s] + dl*DLBYTES);
        c = __builtin_amdgcn_wmma_f32_16x16x32_bf16(
                false, a.v, false, bw[s], (short)0, c, false, false);
      }
      // D: lane col = cout, VGPR i = x0 + xoff + i  -> 8 consecutive x
      if (co16 < COUT) {
        size_t ob = ((size_t)k*COUT + co16)*VOL + (size_t)d*PLANE
                    + y*W_ + x0 + xoff;
        if constexpr (sizeof(OUT_T) == 2) {
          v8bf pk;
          #pragma unroll
          for (int i = 0; i < 8; ++i) {
            float v = c[i] + bias;
            if (RELU) v = v > 0.f ? v : 0.f;
            pk[i] = (__bf16)v;
          }
          *(v8bf*)((__bf16*)Y + ob) = pk;              // global_store_b128
        } else {
          v8f pf;
          #pragma unroll
          for (int i = 0; i < 8; ++i) {
            float v = c[i] + bias;
            if (RELU) v = v > 0.f ? v : 0.f;
            pf[i] = v;
          }
          *(v8f*)((float*)Y + ob) = pf;                // 2x global_store_b128
        }
      }
    }
  }
}

// ---------------------------------------------------------------------------
// Stage 4: softmax over D + expectation (bandwidth-trivial)
// ---------------------------------------------------------------------------
__global__ __launch_bounds__(256)
void softmax_kernel(const float* __restrict__ cost,
                    const float* __restrict__ hypo,
                    float* __restrict__ out) {
  int idx = blockIdx.x * blockDim.x + threadIdx.x;     // (k, y, x)
  if (idx >= K_SAMP * PLANE) return;
  int xy = idx & (PLANE - 1);
  int k = idx >> 12;
  int b = k / F_;
  const float* cp = cost + (size_t)k*VOL + xy;
  float v[D_];
  float mx = -3.0e38f;
  #pragma unroll
  for (int d = 0; d < D_; ++d) {
    v[d] = cp[(size_t)d*PLANE];
    mx = fmaxf(mx, v[d]);
  }
  float sum = 0.f, ex = 0.f;
  #pragma unroll
  for (int d = 0; d < D_; ++d) {
    float e = __expf(v[d] - mx);
    sum += e;
    ex  += e * hypo[b*D_ + d];
  }
  out[idx] = ex / sum;
}

// ---------------------------------------------------------------------------
extern "C" void kernel_launch(void* const* d_in, const int* in_sizes, int n_in,
                              void* d_out, int out_size, void* d_ws, size_t ws_size,
                              hipStream_t stream) {
  const float* features  = (const float*)d_in[0];
  const float* cameras   = (const float*)d_in[1];
  const float* to_worlds = (const float*)d_in[2];
  const float* hypo      = (const float*)d_in[3];
  const float* w0 = (const float*)d_in[4];
  const float* b0 = (const float*)d_in[5];
  const float* w1 = (const float*)d_in[6];
  const float* b1 = (const float*)d_in[7];
  const float* wp = (const float*)d_in[8];
  const float* bp = (const float*)d_in[9];

  char* ws = (char*)d_ws;
  const size_t volBytes = (size_t)K_SAMP * GROUPS_ * VOL * sizeof(__bf16); // 25.2 MB
  float*  cosphi = (float*)ws;
  __bf16* comb   = (__bf16*)(ws + 256);
  __bf16* x0     = (__bf16*)(ws + 256 + volBytes);
  __bf16* x1     = (__bf16*)(ws + 256 + 2*volBytes);
  float*  cost   = (float*) (ws + 256 + 3*volBytes);   // 6.3 MB

  phase_kernel<<<1, 64, 0, stream>>>(cameras, to_worlds, hypo, cosphi);

  int nComb = K_SAMP * GROUPS_ * PLANE;
  combine_kernel<<<(nComb + 255)/256, 256, 0, stream>>>(features, cosphi, comb);

  conv3d_wmma_kernel<BASE_, true,  __bf16><<<K_SAMP*H_, 128, 0, stream>>>(comb, w0, b0, x0);
  conv3d_wmma_kernel<BASE_, true,  __bf16><<<K_SAMP*H_, 128, 0, stream>>>(x0,   w1, b1, x1);
  conv3d_wmma_kernel<1,     false, float ><<<K_SAMP*H_, 128, 0, stream>>>(x1,   wp, bp, cost);

  int nFin = K_SAMP * PLANE;
  softmax_kernel<<<(nFin + 255)/256, 256, 0, stream>>>(cost, hypo, (float*)d_out);
}
